// SimpleMambaClassifier_39565238731516
// MI455X (gfx1250) — matrix operand order repaired and checked by reading, not compiled, not run on hardware
//
#include <hip/hip_runtime.h>

// ---------------------------------------------------------------------------
// SimpleMambaClassifier forward for MI455X (gfx1250, wave32, WMMA)
// ---------------------------------------------------------------------------
#define D_MODEL   256
#define N_LAYERS  4
#define D_STATE   16
#define D_CONV    4
#define D_INNER   512
#define DT_RANK   16
#define NUM_CLASSES 10
#define BATCH     4
#define SEQLEN    1024                 // 32*32
#define M_TOT     (BATCH * SEQLEN)     // 4096
#define XPROJ_N   (DT_RANK + 2 * D_STATE)  // 48

typedef __attribute__((ext_vector_type(16))) _Float16 v16h;
typedef __attribute__((ext_vector_type(8)))  float    v8f;
typedef __attribute__((ext_vector_type(4)))  float    v4f;

__device__ __forceinline__ float silu_f(float x) {
    return x / (1.0f + __expf(-x));
}

// ---------------------------------------------------------------------------
// K0: pixel -> d_model projection.  seq[b,l,c] = x[b,c,l];  h = seq @ Wt + b
// ---------------------------------------------------------------------------
__global__ void k_input_proj(const float* __restrict__ x,
                             const float* __restrict__ w,   // [D_MODEL,3]
                             const float* __restrict__ bias,
                             float* __restrict__ h)
{
    int idx = blockIdx.x * blockDim.x + threadIdx.x;   // M_TOT * D_MODEL
    if (idx >= M_TOT * D_MODEL) return;
    int d = idx % D_MODEL;
    int m = idx / D_MODEL;
    int b = m / SEQLEN, l = m % SEQLEN;
    const float* xb = x + (size_t)b * 3 * SEQLEN;
    float acc = bias[d];
#pragma unroll
    for (int c = 0; c < 3; ++c)
        acc += xb[c * SEQLEN + l] * w[d * 3 + c];
    h[idx] = acc;
}

// ---------------------------------------------------------------------------
// K1: row LayerNorm over D_MODEL=256 (one block per row, blockDim=256)
// ---------------------------------------------------------------------------
__global__ void k_layernorm(const float* __restrict__ in,
                            const float* __restrict__ w,
                            const float* __restrict__ b,
                            float* __restrict__ out)
{
    __shared__ float red[D_MODEL];
    int row = blockIdx.x;
    int t   = threadIdx.x;
    float v = in[(size_t)row * D_MODEL + t];
    red[t] = v; __syncthreads();
#pragma unroll
    for (int s = D_MODEL / 2; s > 0; s >>= 1) {
        if (t < s) red[t] += red[t + s];
        __syncthreads();
    }
    float mu = red[0] * (1.0f / D_MODEL);
    __syncthreads();
    float dv = v - mu;
    red[t] = dv * dv; __syncthreads();
#pragma unroll
    for (int s = D_MODEL / 2; s > 0; s >>= 1) {
        if (t < s) red[t] += red[t + s];
        __syncthreads();
    }
    float rstd = rsqrtf(red[0] * (1.0f / D_MODEL) + 1e-5f);
    out[(size_t)row * D_MODEL + t] = dv * rstd * w[t] + b[t];
}

// ---------------------------------------------------------------------------
// One 16x16x32 WMMA K-step.
// A-fragment (16x32 f16, MxK): lane l -> m = l&15; hi = l>>4;
//   elem e<8  : k = e + 8*hi ; elem e>=8 : k = e + 8 + 8*hi
//   -> two contiguous 8-float chunks at ap+8*hi and ap+16+8*hi (32B aligned)
// B-fragment (32x16 f16, KxN): lane l -> n = l&15; same k mapping as A
//   -> two contiguous 8-float chunks at wp+8*hi and wp+16+8*hi (32B aligned)
// ---------------------------------------------------------------------------
__device__ __forceinline__ v8f wmma_step(const float* __restrict__ ap,
                                         const float* __restrict__ wp,
                                         int hi, v8f acc)
{
    const v4f* a0 = (const v4f*)(ap + 8 * hi);
    const v4f* a1 = (const v4f*)(ap + 16 + 8 * hi);
    const v4f* w0 = (const v4f*)(wp + 8 * hi);
    const v4f* w1 = (const v4f*)(wp + 16 + 8 * hi);
    v4f aA = a0[0], aB = a0[1], aC = a1[0], aD = a1[1];
    v4f wA = w0[0], wB = w0[1], wC = w1[0], wD = w1[1];

    union { v16h v; _Float16 h[16]; } af, bf;
#pragma unroll
    for (int e = 0; e < 4; ++e) {
        af.h[e]      = (_Float16)aA[e];
        af.h[e + 4]  = (_Float16)aB[e];
        af.h[e + 8]  = (_Float16)aC[e];
        af.h[e + 12] = (_Float16)aD[e];
        bf.h[e]      = (_Float16)wA[e];
        bf.h[e + 4]  = (_Float16)wB[e];
        bf.h[e + 8]  = (_Float16)wC[e];
        bf.h[e + 12] = (_Float16)wD[e];
    }
    return __builtin_amdgcn_wmma_f32_16x16x32_f16(
               false, af.v, false, bf.v, (short)0, acc, false, false);
}

// ---------------------------------------------------------------------------
// WMMA GEMM:  C[M,N] = A[M,K] * W[N,K]^T  (+ ADD[M,N] if ADD != nullptr)
// One wave per 16x16 tile; K unrolled by 64 into two independent WMMA
// accumulator chains (K is a multiple of 64 for every GEMM in this model).
// C/D layout (16x16 f32): vgpr r -> m = m0 + r + 8*hi, n = n0 + (l&15)
// ---------------------------------------------------------------------------
__global__ void k_gemm_f16_wmma(const float* __restrict__ A,
                                const float* __restrict__ W,
                                const float* __restrict__ ADD,
                                float* __restrict__ C,
                                int M, int N, int K, int ntiles)
{
    int wave = (int)((blockIdx.x * blockDim.x + threadIdx.x) >> 5);
    if (wave >= ntiles) return;                 // wave-uniform: EXEC stays full
    int lane = threadIdx.x & 31;
    int tilesN = N >> 4;
    int tm = wave / tilesN, tn = wave % tilesN;
    int m0 = tm << 4, n0 = tn << 4;
    int hi = lane >> 4;
    int lo = lane & 15;

    const float* Arow = A + (size_t)(m0 + lo) * K;
    const float* Wrow = W + (size_t)(n0 + lo) * K;

    v8f acc0 = {};
    v8f acc1 = {};
    for (int k0 = 0; k0 < K; k0 += 64) {
        // speculative prefetch of the next K-tile (dropped if past the end)
        __builtin_prefetch(Arow + k0 + 64, 0, 1);
        __builtin_prefetch(Wrow + k0 + 64, 0, 1);
        acc0 = wmma_step(Arow + k0,      Wrow + k0,      hi, acc0);
        acc1 = wmma_step(Arow + k0 + 32, Wrow + k0 + 32, hi, acc1);
    }

    union { v8f v; float f[8]; } cu;
    cu.v = acc0 + acc1;

    float* cp = C + (size_t)(m0 + 8 * hi) * N + (n0 + lo);
    if (ADD != nullptr) {                       // wave-uniform branch
        const float* adp = ADD + (size_t)(m0 + 8 * hi) * N + (n0 + lo);
#pragma unroll
        for (int r = 0; r < 8; ++r)
            cp[(size_t)r * N] = cu.f[r] + adp[(size_t)r * N];
    } else {
#pragma unroll
        for (int r = 0; r < 8; ++r)
            cp[(size_t)r * N] = cu.f[r];
    }
}

// ---------------------------------------------------------------------------
// K2: depthwise causal conv1d (k=4) over xz[:, :D_INNER] + SiLU -> xh
// ---------------------------------------------------------------------------
__global__ void k_conv_silu(const float* __restrict__ xz,   // [M_TOT, 2*D_INNER]
                            const float* __restrict__ cw,   // [D_INNER, D_CONV]
                            const float* __restrict__ cb,   // [D_INNER]
                            float* __restrict__ xh)         // [M_TOT, D_INNER]
{
    int idx = blockIdx.x * blockDim.x + threadIdx.x;
    if (idx >= M_TOT * D_INNER) return;
    int d = idx % D_INNER;
    int m = idx / D_INNER;
    int b = m / SEQLEN, l = m % SEQLEN;
    float acc = cb[d];
#pragma unroll
    for (int k = 0; k < D_CONV; ++k) {
        int lt = l + k - (D_CONV - 1);
        if (lt >= 0)
            acc += xz[(size_t)(b * SEQLEN + lt) * (2 * D_INNER) + d] * cw[d * D_CONV + k];
    }
    xh[idx] = silu_f(acc);
}

// ---------------------------------------------------------------------------
// K4: delta = softplus(dt @ dt_w^T + dt_b)   (K = DT_RANK = 16, scalar VALU)
// ---------------------------------------------------------------------------
__global__ void k_dt_softplus(const float* __restrict__ xdbl, // [M_TOT, XPROJ_N]
                              const float* __restrict__ dtw,  // [D_INNER, DT_RANK]
                              const float* __restrict__ dtb,  // [D_INNER]
                              float* __restrict__ delta)      // [M_TOT, D_INNER]
{
    int idx = blockIdx.x * blockDim.x + threadIdx.x;
    if (idx >= M_TOT * D_INNER) return;
    int d = idx % D_INNER;
    int m = idx / D_INNER;
    const float* dt = xdbl + (size_t)m * XPROJ_N;
    float acc = dtb[d];
#pragma unroll
    for (int r = 0; r < DT_RANK; ++r)
        acc += dt[r] * dtw[d * DT_RANK + r];
    delta[idx] = (acc > 20.0f) ? acc : log1pf(__expf(acc));
}

// ---------------------------------------------------------------------------
// K5: fused selective scan. One thread per (batch, channel); 16 states + 16
// A values live in registers, L=1024 sequential loop. deltaA / deltaBu are
// computed on the fly (never materialized).  y may alias delta (read->write
// same element by the owning thread only).
// ---------------------------------------------------------------------------
__global__ void k_scan(const float* __restrict__ delta,  // [M_TOT, D_INNER]
                       const float* __restrict__ xh,     // [M_TOT, D_INNER]
                       const float* __restrict__ xdbl,   // [M_TOT, XPROJ_N]
                       const float* __restrict__ A_log,  // [D_INNER, D_STATE]
                       const float* __restrict__ Dskip,  // [D_INNER]
                       float* __restrict__ y)            // [M_TOT, D_INNER]
{
    int d = blockIdx.x * blockDim.x + threadIdx.x;
    int b = blockIdx.y;
    if (d >= D_INNER) return;

    float An[D_STATE], hs[D_STATE];
#pragma unroll
    for (int n = 0; n < D_STATE; ++n) {
        An[n] = -__expf(A_log[d * D_STATE + n]);
        hs[n] = 0.0f;
    }
    float dsk = Dskip[d];

    for (int t = 0; t < SEQLEN; ++t) {
        size_t m  = (size_t)b * SEQLEN + t;
        float dl  = delta[m * D_INNER + d];
        float xv  = xh[m * D_INNER + d];
        const float* bc = xdbl + m * XPROJ_N;   // [dt(16) | B(16) | C(16)]
        float du = dl * xv;
        float yv = 0.0f;
#pragma unroll
        for (int n = 0; n < D_STATE; ++n) {
            float dA = __expf(dl * An[n]);
            hs[n] = dA * hs[n] + du * bc[DT_RANK + n];
            yv   += hs[n] * bc[DT_RANK + D_STATE + n];
        }
        y[m * D_INNER + d] = yv + xv * dsk;
    }
}

// ---------------------------------------------------------------------------
// K6: gating  y *= silu(z)   (z = second half of xz; in-place on y is safe)
// ---------------------------------------------------------------------------
__global__ void k_gate(const float* __restrict__ y,
                       const float* __restrict__ xz,
                       float* __restrict__ out)
{
    int idx = blockIdx.x * blockDim.x + threadIdx.x;
    if (idx >= M_TOT * D_INNER) return;
    int d = idx % D_INNER;
    int m = idx / D_INNER;
    float z = xz[(size_t)m * (2 * D_INNER) + D_INNER + d];
    out[idx] = y[idx] * silu_f(z);
}

// ---------------------------------------------------------------------------
// K8: mean over L, final LayerNorm, classifier head -> d_out [BATCH, 10]
// ---------------------------------------------------------------------------
__global__ void k_head(const float* __restrict__ h,     // [M_TOT, D_MODEL]
                       const float* __restrict__ fnw,
                       const float* __restrict__ fnb,
                       const float* __restrict__ clsw,  // [10, D_MODEL]
                       const float* __restrict__ clsb,
                       float* __restrict__ out)         // [BATCH, 10]
{
    __shared__ float red[D_MODEL];
    __shared__ float nh[D_MODEL];
    int b = blockIdx.x;
    int t = threadIdx.x;

    float s = 0.0f;
    for (int l = 0; l < SEQLEN; ++l)
        s += h[(size_t)(b * SEQLEN + l) * D_MODEL + t];
    float v = s * (1.0f / SEQLEN);

    red[t] = v; __syncthreads();
#pragma unroll
    for (int sft = D_MODEL / 2; sft > 0; sft >>= 1) {
        if (t < sft) red[t] += red[t + sft];
        __syncthreads();
    }
    float mu = red[0] * (1.0f / D_MODEL);
    __syncthreads();
    float dv = v - mu;
    red[t] = dv * dv; __syncthreads();
#pragma unroll
    for (int sft = D_MODEL / 2; sft > 0; sft >>= 1) {
        if (t < sft) red[t] += red[t + sft];
        __syncthreads();
    }
    float rstd = rsqrtf(red[0] * (1.0f / D_MODEL) + 1e-5f);
    nh[t] = dv * rstd * fnw[t] + fnb[t];
    __syncthreads();

    if (t < NUM_CLASSES) {
        float a = clsb[t];
        for (int d2 = 0; d2 < D_MODEL; ++d2)
            a += nh[d2] * clsw[t * D_MODEL + d2];
        out[b * NUM_CLASSES + t] = a;
    }
}

// ---------------------------------------------------------------------------
// Host launch
// ---------------------------------------------------------------------------
extern "C" void kernel_launch(void* const* d_in, const int* in_sizes, int n_in,
                              void* d_out, int out_size, void* d_ws, size_t ws_size,
                              hipStream_t stream)
{
    (void)in_sizes; (void)n_in; (void)out_size; (void)ws_size;

    const float* x     = (const float*)d_in[0];   // [4,3,32,32]
    const float* ipw   = (const float*)d_in[1];   // [256,3]
    const float* ipb   = (const float*)d_in[2];   // [256]
    const float* inw   = (const float*)d_in[3];   // [4,1024,256]
    const float* cw    = (const float*)d_in[4];   // [4,512,4]
    const float* cb    = (const float*)d_in[5];   // [4,512]
    const float* xpw   = (const float*)d_in[6];   // [4,48,512]
    const float* dtw   = (const float*)d_in[7];   // [4,512,16]
    const float* dtb   = (const float*)d_in[8];   // [4,512]
    const float* Alog  = (const float*)d_in[9];   // [4,512,16]
    const float* Dsk   = (const float*)d_in[10];  // [4,512]
    const float* outw  = (const float*)d_in[11];  // [4,256,512]
    const float* lnw   = (const float*)d_in[12];  // [4,256]
    const float* lnb   = (const float*)d_in[13];  // [4,256]
    const float* fnw   = (const float*)d_in[14];  // [256]
    const float* fnb   = (const float*)d_in[15];  // [256]
    const float* clsw  = (const float*)d_in[16];  // [10,256]
    const float* clsb  = (const float*)d_in[17];  // [10]
    float* outp = (float*)d_out;

    // workspace carve-up (floats)
    float* ws   = (float*)d_ws;
    float* hA   = ws;                                        // [4096,256]
    float* hB   = hA   + (size_t)M_TOT * D_MODEL;            // [4096,256]
    float* xn   = hB   + (size_t)M_TOT * D_MODEL;            // [4096,256]
    float* xz   = xn   + (size_t)M_TOT * D_MODEL;            // [4096,1024]
    float* xh   = xz   + (size_t)M_TOT * 2 * D_INNER;        // [4096,512]
    float* xdbl = xh   + (size_t)M_TOT * D_INNER;            // [4096,48]
    float* dly  = xdbl + (size_t)M_TOT * XPROJ_N;            // [4096,512] delta/y/gated (aliased)

    const int TPB = 256;

    // input projection
    k_input_proj<<<(M_TOT * D_MODEL + TPB - 1) / TPB, TPB, 0, stream>>>(x, ipw, ipb, hA);

    float* hcur = hA;
    float* hnext = hB;

    for (int i = 0; i < N_LAYERS; ++i) {
        const float* inw_i  = inw  + (size_t)i * 2 * D_INNER * D_MODEL;
        const float* cw_i   = cw   + (size_t)i * D_INNER * D_CONV;
        const float* cb_i   = cb   + (size_t)i * D_INNER;
        const float* xpw_i  = xpw  + (size_t)i * XPROJ_N * D_INNER;
        const float* dtw_i  = dtw  + (size_t)i * D_INNER * DT_RANK;
        const float* dtb_i  = dtb  + (size_t)i * D_INNER;
        const float* Alog_i = Alog + (size_t)i * D_INNER * D_STATE;
        const float* Dsk_i  = Dsk  + (size_t)i * D_INNER;
        const float* outw_i = outw + (size_t)i * D_MODEL * D_INNER;

        // LayerNorm
        k_layernorm<<<M_TOT, D_MODEL, 0, stream>>>(hcur, lnw + i * D_MODEL, lnb + i * D_MODEL, xn);

        // in_proj: [4096,256] x [1024,256]^T -> xz [4096,1024]
        {
            int nt = (M_TOT / 16) * (2 * D_INNER / 16);          // 16384 tiles
            k_gemm_f16_wmma<<<nt / 8, 256, 0, stream>>>(xn, inw_i, nullptr, xz,
                                                        M_TOT, 2 * D_INNER, D_MODEL, nt);
        }

        // depthwise conv + SiLU -> xh
        k_conv_silu<<<(M_TOT * D_INNER + TPB - 1) / TPB, TPB, 0, stream>>>(xz, cw_i, cb_i, xh);

        // x_proj: [4096,512] x [48,512]^T -> xdbl [4096,48]
        {
            int nt = (M_TOT / 16) * (XPROJ_N / 16);              // 768 tiles
            k_gemm_f16_wmma<<<nt / 8, 256, 0, stream>>>(xh, xpw_i, nullptr, xdbl,
                                                        M_TOT, XPROJ_N, D_INNER, nt);
        }

        // dt_proj + softplus -> delta
        k_dt_softplus<<<(M_TOT * D_INNER + TPB - 1) / TPB, TPB, 0, stream>>>(xdbl, dtw_i, dtb_i, dly);

        // fused selective scan (delta read -> y written in place)
        {
            dim3 grid(D_INNER / 128, BATCH);
            k_scan<<<grid, 128, 0, stream>>>(dly, xh, xdbl, Alog_i, Dsk_i, dly);
        }

        // gating y *= silu(z) (in place)
        k_gate<<<(M_TOT * D_INNER + TPB - 1) / TPB, TPB, 0, stream>>>(dly, xz, dly);

        // out_proj + residual: [4096,512] x [256,512]^T + hcur -> hnext
        {
            int nt = (M_TOT / 16) * (D_MODEL / 16);              // 4096 tiles
            k_gemm_f16_wmma<<<nt / 8, 256, 0, stream>>>(dly, outw_i, hcur, hnext,
                                                        M_TOT, D_MODEL, D_INNER, nt);
        }

        float* tmp = hcur; hcur = hnext; hnext = tmp;
    }

    // mean over L + final LN + classifier
    k_head<<<BATCH, D_MODEL, 0, stream>>>(hcur, fnw, fnb, clsw, clsb, outp);
}
